// DGCNN_50062138802927
// MI455X (gfx1250) — compile-verified, matrix operand order checked by reading, not run on hardware
//
#include <hip/hip_runtime.h>

#define NPTS 8192
#define NT   (NPTS / 16)   // 512 n-tiles

typedef __bf16 bf16;
typedef bf16  v8bf  __attribute__((ext_vector_type(8)));
typedef bf16  v16bf __attribute__((ext_vector_type(16)));
typedef float v8f   __attribute__((ext_vector_type(8)));

static __device__ inline bf16 f2bf(float f) {
    unsigned u = __builtin_bit_cast(unsigned, f);
    u += 0x7FFFu + ((u >> 16) & 1u);          // round-to-nearest-even
    unsigned short h = (unsigned short)(u >> 16);
    return __builtin_bit_cast(bf16, h);
}
static __device__ inline float bf2f(bf16 b) {
    unsigned u = (unsigned)__builtin_bit_cast(unsigned short, b) << 16;
    return __builtin_bit_cast(float, u);
}
static __device__ inline v16bf cat8(v8bf lo, v8bf hi) {
    return __builtin_shufflevector(lo, hi, 0,1,2,3,4,5,6,7,8,9,10,11,12,13,14,15);
}
static __device__ inline v16bf ldfrag(const bf16* p) {
    return cat8(*(const v8bf*)p, *(const v8bf*)(p + 8));
}
// B-fragment-native blocked layout index for element (k, n):
// Xblk[((kc*NT + nt)*32 + lane)*16 + e], lane = hf*16 + n%16
static __device__ inline size_t blk_idx(int k, int n) {
    int kc = k >> 5, kr = k & 31;
    int nt = n >> 4, l = n & 15;
    int hf, e;
    if (kr < 16) { hf = kr >> 3; e = kr & 7; }
    else         { hf = (kr - 16) >> 3; e = 8 + ((kr - 16) & 7); }
    return ((size_t)(kc * NT + nt) * 32 + hf * 16 + l) * 16 + e;
}

// ---------------------------------------------------------------------------
// Bake weights into WMMA A-fragment order (bf16), zero-padded to Kpad and
// Mtiles*16. wdiff=1 bakes W[:,K:]-W[:,:K].  Apk[((mo*Kc+kc)*32+lane)*16+j]
// ---------------------------------------------------------------------------
__global__ void pack_w(const float* __restrict__ W, int ldw, int Kreal, int Kpad,
                       int O, int wdiff, bf16* __restrict__ out)
{
    const int Kc = Kpad >> 5;
    const int Mtiles = (O + 15) >> 4;
    const int t = blockIdx.x * 256 + threadIdx.x;
    if (t >= Mtiles * Kc * 32) return;
    const int lane = t & 31;
    const int kc   = (t >> 5) % Kc;
    const int mo   = t / (32 * Kc);
    const int hf = lane >> 4, l = lane & 15;
    const int m = mo * 16 + l;
    bf16* op = out + (size_t)t * 16;
#pragma unroll
    for (int j = 0; j < 16; ++j) {
        int kg = kc * 32 + ((j < 8) ? (hf * 8 + j) : (16 + hf * 8 + (j - 8)));
        float v = 0.f;
        if (m < O && kg < Kreal)
            v = wdiff ? (W[(size_t)m * ldw + Kreal + kg] - W[(size_t)m * ldw + kg])
                      : (W[(size_t)m * ldw + kg]);
        op[j] = f2bf(v);
    }
}

// ---------------------------------------------------------------------------
// D = act((Apk * X) * g + b). Xblk in fragment-blocked bf16 layout (K x NPTS).
// 8 waves / 256-thread block, barrier-free: each wave one 16(M) x 32(N) tile.
// Software-pipelined: fragments of chunk kc+1 are in flight during chunk kc's
// WMMAs, so waits are partial instead of s_wait_loadcnt 0.
// omode: 1 = f32 (NPTS,O) transposed; 2 = bf16 fragment-blocked (O x NPTS)
// ---------------------------------------------------------------------------
__global__ void gemm_wmma(const bf16* __restrict__ Apk,
                          const bf16* __restrict__ Xblk, int K,
                          const float* __restrict__ g, const float* __restrict__ bia,
                          float* __restrict__ outF, bf16* __restrict__ outH,
                          int O, int relu, int omode)
{
    const int lane = threadIdx.x & 31;
    const int wv   = threadIdx.x >> 5;
    const int hf = lane >> 4, l = lane & 15;
    const int Mtiles = (O + 15) >> 4;
    const int mo = blockIdx.y * 8 + wv;
    if (mo >= Mtiles) return;
    const int Kc = K >> 5;
    const int nt0 = blockIdx.x * 2;

    v8f acc0 = {0.f,0.f,0.f,0.f,0.f,0.f,0.f,0.f};
    v8f acc1 = {0.f,0.f,0.f,0.f,0.f,0.f,0.f,0.f};

    const bf16* ap  = Apk  + ((size_t)mo * Kc * 32 + lane) * 16;
    const bf16* bp0 = Xblk + ((size_t)nt0 * 32 + lane) * 16;
    const bf16* bp1 = bp0 + 512;                       // next n-tile
    const size_t bstep = (size_t)NT * 512;             // next K-chunk

    v16bf a  = ldfrag(ap);
    v16bf b0 = ldfrag(bp0);
    v16bf b1 = ldfrag(bp1);

    for (int kc = 0; kc < Kc - 1; ++kc) {
        ap += 512; bp0 += bstep; bp1 += bstep;
        v16bf an  = ldfrag(ap);                        // in flight during wmma
        v16bf b0n = ldfrag(bp0);
        v16bf b1n = ldfrag(bp1);
        if (kc + 2 < Kc) {                             // warm L2 at distance 2
            __builtin_prefetch(ap + 512, 0, 1);
            __builtin_prefetch(bp0 + bstep, 0, 1);
            __builtin_prefetch(bp1 + bstep, 0, 1);
        }
        acc0 = __builtin_amdgcn_wmma_f32_16x16x32_bf16(false, a, false, b0,
                                                       (short)0, acc0, false, false);
        acc1 = __builtin_amdgcn_wmma_f32_16x16x32_bf16(false, a, false, b1,
                                                       (short)0, acc1, false, false);
        a = an; b0 = b0n; b1 = b1n;
    }
    acc0 = __builtin_amdgcn_wmma_f32_16x16x32_bf16(false, a, false, b0,
                                                   (short)0, acc0, false, false);
    acc1 = __builtin_amdgcn_wmma_f32_16x16x32_bf16(false, a, false, b1,
                                                   (short)0, acc1, false, false);
#pragma unroll
    for (int s = 0; s < 2; ++s) {
        const int n = nt0 * 16 + s * 16 + l;
#pragma unroll
        for (int r = 0; r < 8; ++r) {
            const int mm = mo * 16 + r + hf * 8;
            if (mm < O) {
                float v = s ? acc1[r] : acc0[r];
                float gg = g   ? g[mm]   : 1.f;
                float bb = bia ? bia[mm] : 0.f;
                v = v * gg + bb;
                if (relu) v = v > 0.f ? v : 0.2f * v;
                if (omode == 1) outF[(size_t)n * O + mm] = v;
                else            outH[blk_idx(mm, n)] = f2bf(v);
            }
        }
    }
}

// ---------------------------------------------------------------------------
// KNN top-20: pdist = 2*<xi,xj> - |xi|^2 - |xj|^2 (larger = closer).
// One wave per 16 queries. XT (NPTS x CP) bf16 row-major (A side),
// Xblk fragment-blocked (B side). CP % 32 == 0.
// Next tile's B fragments are loaded before the top-20 scan so the gather
// latency overlaps the VALU scan phase.
// ---------------------------------------------------------------------------
template <int CP>
__global__ void knn_wmma(const bf16* __restrict__ XT, const bf16* __restrict__ Xblk,
                         const float* __restrict__ sq, int* __restrict__ idx)
{
    __shared__ float ldsD[16 * 16];
    const int lane = threadIdx.x & 31;
    const int hf = lane >> 4, l = lane & 15;
    const int q0 = blockIdx.x * 16;
    constexpr int Kc = CP / 32;

    v16bf afr[Kc];
    const bf16* xr = XT + (size_t)(q0 + l) * CP;
#pragma unroll
    for (int kc = 0; kc < Kc; ++kc) {
        const bf16* p1 = xr + kc * 32 + hf * 8;
        afr[kc] = cat8(*(const v8bf*)p1, *(const v8bf*)(p1 + 16));
    }
    float sqq[8];
#pragma unroll
    for (int r = 0; r < 8; ++r) sqq[r] = sq[q0 + r + hf * 8];

    float bd[20]; int bi[20];
#pragma unroll
    for (int s = 0; s < 20; ++s) { bd[s] = -3.4e38f; bi[s] = 0; }

    const bf16* bbase = Xblk + (size_t)lane * 16;
    v16bf bcur[Kc];
#pragma unroll
    for (int kc = 0; kc < Kc; ++kc)
        bcur[kc] = ldfrag(bbase + (size_t)kc * NT * 512);

    for (int ct = 0; ct < NT; ++ct) {
        v8f acc = {0.f,0.f,0.f,0.f,0.f,0.f,0.f,0.f};
#pragma unroll
        for (int kc = 0; kc < Kc; ++kc)
            acc = __builtin_amdgcn_wmma_f32_16x16x32_bf16(false, afr[kc], false,
                                                          bcur[kc], (short)0, acc,
                                                          false, false);
        // issue next tile's fragment loads; they complete during the scan
        v16bf bnext[Kc];
        if (ct + 1 < NT) {
#pragma unroll
            for (int kc = 0; kc < Kc; ++kc)
                bnext[kc] = ldfrag(bbase + (size_t)(kc * NT + ct + 1) * 512);
            if (ct + 2 < NT)
                __builtin_prefetch(bbase + (size_t)(ct + 2) * 512, 0, 1);
        }

        const int n0 = ct * 16;
        const float sqc = sq[n0 + l];
#pragma unroll
        for (int r = 0; r < 8; ++r)
            ldsD[(r + hf * 8) * 16 + l] = 2.f * acc[r] - sqq[r] - sqc;
        __syncthreads();
        if (lane < 16) {
#pragma unroll
            for (int j = 0; j < 16; ++j) {
                float v = ldsD[lane * 16 + j];
                if (v > bd[19]) {
                    float cv = v; int ci = n0 + j;
#pragma unroll
                    for (int s = 0; s < 20; ++s) {
                        if (cv > bd[s]) {
                            float td = bd[s]; int ti = bi[s];
                            bd[s] = cv; bi[s] = ci; cv = td; ci = ti;
                        }
                    }
                }
            }
        }
        __syncthreads();
        if (ct + 1 < NT) {
#pragma unroll
            for (int kc = 0; kc < Kc; ++kc) bcur[kc] = bnext[kc];
        }
    }
    if (lane < 16) {
#pragma unroll
        for (int s = 0; s < 20; ++s)
            idx[(size_t)(q0 + lane) * 20 + s] = bi[s];
    }
}

// ---------------------------------------------------------------------------
// EdgeConv epilogue: out[o][n] = max_k lrelu((y[idx[n][k]][o] + z[n][o])*g + b)
// yT/zT f32 (NPTS x O). Outputs: fragment-blocked bf16 + row-major (NPTS x O).
// ---------------------------------------------------------------------------
__global__ void gather_max(const float* __restrict__ yT, const float* __restrict__ zT,
                           const int* __restrict__ idx, const float* __restrict__ g,
                           const float* __restrict__ bia, int O,
                           bf16* __restrict__ Xblk, bf16* __restrict__ XT)
{
    const int t  = blockIdx.x * blockDim.x + threadIdx.x;
    const int o4 = O >> 2;
    const int n  = t / o4;
    const int o  = (t % o4) * 4;
    if (n >= NPTS) return;

    float4 z4 = *(const float4*)(zT + (size_t)n * O + o);
    float4 g4 = *(const float4*)(g + o);
    float4 b4 = *(const float4*)(bia + o);
    float4 mx = make_float4(-3.4e38f, -3.4e38f, -3.4e38f, -3.4e38f);
    const int* ir = idx + (size_t)n * 20;
#pragma unroll 4
    for (int k = 0; k < 20; ++k) {
        int j = ir[k];
        float4 y4 = *(const float4*)(yT + (size_t)j * O + o);
        float vx = (y4.x + z4.x) * g4.x + b4.x; vx = vx > 0.f ? vx : 0.2f * vx;
        float vy = (y4.y + z4.y) * g4.y + b4.y; vy = vy > 0.f ? vy : 0.2f * vy;
        float vz = (y4.z + z4.z) * g4.z + b4.z; vz = vz > 0.f ? vz : 0.2f * vz;
        float vw = (y4.w + z4.w) * g4.w + b4.w; vw = vw > 0.f ? vw : 0.2f * vw;
        mx.x = fmaxf(mx.x, vx); mx.y = fmaxf(mx.y, vy);
        mx.z = fmaxf(mx.z, vz); mx.w = fmaxf(mx.w, vw);
    }
    XT[(size_t)n * O + o + 0] = f2bf(mx.x);
    XT[(size_t)n * O + o + 1] = f2bf(mx.y);
    XT[(size_t)n * O + o + 2] = f2bf(mx.z);
    XT[(size_t)n * O + o + 3] = f2bf(mx.w);
    Xblk[blk_idx(o + 0, n)] = f2bf(mx.x);
    Xblk[blk_idx(o + 1, n)] = f2bf(mx.y);
    Xblk[blk_idx(o + 2, n)] = f2bf(mx.z);
    Xblk[blk_idx(o + 3, n)] = f2bf(mx.w);
}

// |x_n|^2 from row-major XT (NPTS x C), C % 8 == 0 (zero-padded ok)
__global__ void sq_norm(const bf16* __restrict__ XT, int C, float* __restrict__ sq)
{
    int n = blockIdx.x * blockDim.x + threadIdx.x;
    if (n >= NPTS) return;
    const bf16* xr = XT + (size_t)n * C;
    float s = 0.f;
    for (int c = 0; c < C; c += 8) {
        v8bf v = *(const v8bf*)(xr + c);
#pragma unroll
        for (int j = 0; j < 8; ++j) { float f = bf2f(v[j]); s += f * f; }
    }
    sq[n] = s;
}

// theta_dis (N,3) f32 -> blocked bf16 (32,NPTS) zero-padded + XT (NPTS,32)
__global__ void transpose3(const float* __restrict__ in, bf16* __restrict__ x0blk,
                           bf16* __restrict__ x0T)
{
    int n = blockIdx.x * blockDim.x + threadIdx.x;
    if (n >= NPTS) return;
    const bf16 zero = f2bf(0.f);
#pragma unroll
    for (int c = 0; c < 32; ++c) {
        bf16 v = (c < 3) ? f2bf(in[n * 3 + c]) : zero;
        x0blk[blk_idx(c, n)]     = v;
        x0T[(size_t)n * 32 + c]  = v;
    }
}

// ---------------------------------------------------------------------------
extern "C" void kernel_launch(void* const* d_in, const int* in_sizes, int n_in,
                              void* d_out, int out_size, void* d_ws, size_t ws_size,
                              hipStream_t stream)
{
    (void)in_sizes; (void)n_in; (void)out_size; (void)ws_size;
    const int N = NPTS;
    const float* th  = (const float*)d_in[0];
    const float* W1  = (const float*)d_in[1];
    const float* g1  = (const float*)d_in[2];
    const float* b1  = (const float*)d_in[3];
    const float* W2  = (const float*)d_in[4];
    const float* g2  = (const float*)d_in[5];
    const float* b2  = (const float*)d_in[6];
    const float* W3  = (const float*)d_in[7];
    const float* g3  = (const float*)d_in[8];
    const float* b3  = (const float*)d_in[9];
    const float* W4  = (const float*)d_in[10];
    const float* g4  = (const float*)d_in[11];
    const float* b4  = (const float*)d_in[12];
    const float* W5  = (const float*)d_in[13];
    const float* g5  = (const float*)d_in[14];
    const float* b5  = (const float*)d_in[15];
    const float* L1w = (const float*)d_in[16];
    const float* L1b = (const float*)d_in[17];
    const float* L2w = (const float*)d_in[18];
    const float* L2b = (const float*)d_in[19];
    const float* L3w = (const float*)d_in[20];
    const float* L3b = (const float*)d_in[21];

    char* ws = (char*)d_ws;
    size_t off = 0;
    auto allocB = [&](size_t bytes) {
        void* p = ws + off;
        off += (bytes + 255) & ~(size_t)255;
        return p;
    };
    auto bfa = [&](size_t n_) { return (bf16*)allocB(n_ * 2); };
    auto f32 = [&](size_t n_) { return (float*)allocB(n_ * 4); };

    bf16* x0blk = bfa((size_t)32 * N);
    bf16* x0T   = bfa((size_t)N * 32);
    bf16* xc    = bfa((size_t)512 * N);        // blocked, kc-major: concat for free
    bf16* X1blk = xc;                          // kc 0..1
    bf16* X2blk = xc + (size_t)64  * N;        // kc 2..3
    bf16* X3blk = xc + (size_t)128 * N;        // kc 4..7
    bf16* X4blk = xc + (size_t)256 * N;        // kc 8..15
    bf16* X1T  = bfa((size_t)N * 64);
    bf16* X2T  = bfa((size_t)N * 64);
    bf16* X3T  = bfa((size_t)N * 128);
    bf16* X4T  = bfa((size_t)N * 256);
    float* yT  = f32((size_t)N * 256);
    float* zT  = f32((size_t)N * 256);
    bf16* x5   = bfa((size_t)1024 * N);
    bf16* h1   = bfa((size_t)512  * N);
    bf16* h2   = bfa((size_t)256  * N);
    float* sqb = f32((size_t)N);
    int*  idb  = (int*)allocB((size_t)N * 20 * 4);
    bf16* P1n = bfa((size_t)64 * 32);
    bf16* P1d = bfa((size_t)64 * 32);
    bf16* P2n = bfa((size_t)64 * 64);
    bf16* P2d = bfa((size_t)64 * 64);
    bf16* P3n = bfa((size_t)128 * 64);
    bf16* P3d = bfa((size_t)128 * 64);
    bf16* P4n = bfa((size_t)256 * 128);
    bf16* P4d = bfa((size_t)256 * 128);
    bf16* P5  = bfa((size_t)1024 * 512);
    bf16* PL1 = bfa((size_t)512 * 1024);
    bf16* PL2 = bfa((size_t)256 * 512);
    bf16* PL3 = bfa((size_t)48 * 256);

    const dim3 b32(32), b256(256);
    const int ntg = N / 32;                  // gemm grid.x (32-wide tiles)
    const int ntk = N / 16;                  // knn query tiles
    auto gy = [](int O) { return (((O + 15) / 16) + 7) / 8; };
    auto pk = [&](const float* W, int ldw, int Kr, int Kp, int O, int wd, bf16* out) {
        int total = ((O + 15) / 16) * (Kp / 32) * 32;
        pack_w<<<(total + 255) / 256, b256, 0, stream>>>(W, ldw, Kr, Kp, O, wd, out);
    };

    // ---- bake weights ----
    pk(W1, 6, 3, 32, 64, 0, P1n);       pk(W1, 6, 3, 32, 64, 1, P1d);
    pk(W2, 128, 64, 64, 64, 0, P2n);    pk(W2, 128, 64, 64, 64, 1, P2d);
    pk(W3, 128, 64, 64, 128, 0, P3n);   pk(W3, 128, 64, 64, 128, 1, P3d);
    pk(W4, 256, 128, 128, 256, 0, P4n); pk(W4, 256, 128, 128, 256, 1, P4d);
    pk(W5, 512, 512, 512, 1024, 0, P5);
    pk(L1w, 1024, 1024, 1024, 512, 0, PL1);
    pk(L2w, 512, 512, 512, 256, 0, PL2);
    pk(L3w, 256, 256, 256, 40, 0, PL3);

    // ---- layer 1 (3 -> 64), data zero-padded to K=32 ----
    transpose3<<<N / 256, b256, 0, stream>>>(th, x0blk, x0T);
    sq_norm<<<N / 256, b256, 0, stream>>>(x0T, 32, sqb);
    knn_wmma<32><<<ntk, b32, 0, stream>>>(x0T, x0blk, sqb, idb);
    gemm_wmma<<<dim3(ntg, gy(64)), b256, 0, stream>>>(P1n, x0blk, 32, nullptr, nullptr, yT, nullptr, 64, 0, 1);
    gemm_wmma<<<dim3(ntg, gy(64)), b256, 0, stream>>>(P1d, x0blk, 32, nullptr, nullptr, zT, nullptr, 64, 0, 1);
    gather_max<<<(N * 16) / 256, b256, 0, stream>>>(yT, zT, idb, g1, b1, 64, X1blk, X1T);

    // ---- layer 2 (64 -> 64) ----
    sq_norm<<<N / 256, b256, 0, stream>>>(X1T, 64, sqb);
    knn_wmma<64><<<ntk, b32, 0, stream>>>(X1T, X1blk, sqb, idb);
    gemm_wmma<<<dim3(ntg, gy(64)), b256, 0, stream>>>(P2n, X1blk, 64, nullptr, nullptr, yT, nullptr, 64, 0, 1);
    gemm_wmma<<<dim3(ntg, gy(64)), b256, 0, stream>>>(P2d, X1blk, 64, nullptr, nullptr, zT, nullptr, 64, 0, 1);
    gather_max<<<(N * 16) / 256, b256, 0, stream>>>(yT, zT, idb, g2, b2, 64, X2blk, X2T);

    // ---- layer 3 (64 -> 128) ----
    sq_norm<<<N / 256, b256, 0, stream>>>(X2T, 64, sqb);
    knn_wmma<64><<<ntk, b32, 0, stream>>>(X2T, X2blk, sqb, idb);
    gemm_wmma<<<dim3(ntg, gy(128)), b256, 0, stream>>>(P3n, X2blk, 64, nullptr, nullptr, yT, nullptr, 128, 0, 1);
    gemm_wmma<<<dim3(ntg, gy(128)), b256, 0, stream>>>(P3d, X2blk, 64, nullptr, nullptr, zT, nullptr, 128, 0, 1);
    gather_max<<<(N * 32) / 256, b256, 0, stream>>>(yT, zT, idb, g3, b3, 128, X3blk, X3T);

    // ---- layer 4 (128 -> 256) ----
    sq_norm<<<N / 256, b256, 0, stream>>>(X3T, 128, sqb);
    knn_wmma<128><<<ntk, b32, 0, stream>>>(X3T, X3blk, sqb, idb);
    gemm_wmma<<<dim3(ntg, gy(256)), b256, 0, stream>>>(P4n, X3blk, 128, nullptr, nullptr, yT, nullptr, 256, 0, 1);
    gemm_wmma<<<dim3(ntg, gy(256)), b256, 0, stream>>>(P4d, X3blk, 128, nullptr, nullptr, zT, nullptr, 256, 0, 1);
    gather_max<<<(N * 64) / 256, b256, 0, stream>>>(yT, zT, idb, g4, b4, 256, X4blk, X4T);

    // ---- head: xc already contiguous blocked (512,N) ----
    gemm_wmma<<<dim3(ntg, gy(1024)), b256, 0, stream>>>(P5, xc, 512, g5, b5, nullptr, x5, 1024, 1, 2);
    gemm_wmma<<<dim3(ntg, gy(512)),  b256, 0, stream>>>(PL1, x5, 1024, nullptr, L1b, nullptr, h1, 512, 1, 2);
    gemm_wmma<<<dim3(ntg, gy(256)),  b256, 0, stream>>>(PL2, h1, 512, nullptr, L2b, nullptr, h2, 256, 1, 2);
    // final: (N,40) f32 transposed store, bias only, no activation
    gemm_wmma<<<dim3(ntg, gy(40)),   b256, 0, stream>>>(PL3, h2, 256, nullptr, L3b,
                                                        (float*)d_out, nullptr, 40, 0, 1);
}
